// CLIPTextModelForPromptTuning_4939212390703
// MI455X (gfx1250) — matrix operand order repaired and checked by reading, not compiled, not run on hardware
//
#include <hip/hip_runtime.h>
#include <hip/hip_bf16.h>
#include <math.h>

// ---------------- model constants (from reference) ----------------
#define NB   128     // batch
#define NT   69      // text tokens
#define NNP  8       // prompt tokens
#define NS   77      // sequence after prompt insertion (1 + 8 + 68)
#define ND   512     // model dim
#define NH   8       // heads
#define HD   64      // head dim
#define NFF  2048    // ffn dim
#define NLAY 12
#define DGc  6
#define DSc  6
#define SP   80      // padded sequence (tiles of 16)
#define SPK  96      // padded K for PV wmma (multiple of 32)
#define LROW 72      // padded LDS row stride (halves) -> conflict-free b128 reads
#define NEGV (-1.0e30f)

typedef __attribute__((ext_vector_type(16))) _Float16 v16h;
typedef __attribute__((ext_vector_type(8)))  float    v8f;

union Frag {
    v16h     v;
    _Float16 h[16];
    unsigned u[8];
};

// ---------------- f32 [K,N] -> f16 transposed [N,K] ----------------
__global__ void k_cvt_t(const float* __restrict__ src, _Float16* __restrict__ dst,
                        int K, int N) {
    long i = (long)blockIdx.x * blockDim.x + threadIdx.x;
    long total = (long)K * N;
    if (i >= total) return;
    int n = (int)(i % N);
    int k = (int)(i / N);
    dst[(long)n * K + k] = (_Float16)src[i];
}

// ---------------- embedding + prompt0 + pos ----------------
__global__ void k_embed(const int* __restrict__ tok, const float* __restrict__ emb,
                        const float* __restrict__ pos, const float* __restrict__ gpr,
                        float* __restrict__ X) {
    long i = (long)blockIdx.x * blockDim.x + threadIdx.x;
    long total = (long)NB * NS * ND;
    if (i >= total) return;
    int d = (int)(i % ND);
    long r = i / ND;
    int s = (int)(r % NS);
    int b = (int)(r / NS);
    float v;
    if (s == 0)            v = emb[(long)tok[b * NT] * ND + d];
    else if (s <= NNP)     v = gpr[(((long)b * DGc + 0) * NNP + (s - 1)) * ND + d];
    else                   v = emb[(long)tok[b * NT + (s - NNP)] * ND + d];
    X[i] = v + pos[(long)s * ND + d];
}

// ---------------- replace prompt rows for layer i>0 (no pos emb) ----------------
__global__ void k_set_prompt(const float* __restrict__ gpr, const float* __restrict__ spr,
                             float* __restrict__ X, int layer) {
    int i = blockIdx.x * blockDim.x + threadIdx.x;
    int total = NB * NNP * ND;
    if (i >= total) return;
    int d = i % ND;
    int r = i / ND;
    int p = r % NNP;
    int b = r / NNP;
    float v = (layer < DGc)
        ? gpr[(((long)b * DGc + layer) * NNP + p) * ND + d]
        : spr[(((long)b * DSc + (layer - (NLAY - DSc))) * NNP + p) * ND + d];
    X[((long)b * NS + 1 + p) * ND + d] = v;
}

// ---------------- LayerNorm: f32 row -> f16 row (one wave per row) ----------------
__global__ void k_ln(const float* __restrict__ X, const float* __restrict__ g,
                     const float* __restrict__ bt, _Float16* __restrict__ Hout, int rows) {
    int wid = threadIdx.x >> 5, lane = threadIdx.x & 31;
    int row = blockIdx.x * 8 + wid;
    if (row >= rows) return;
    const float* xr = X + (long)row * ND;
    float s = 0.f;
    for (int i = lane; i < ND; i += 32) s += xr[i];
    #pragma unroll
    for (int o = 16; o; o >>= 1) s += __shfl_xor(s, o, 32);
    float mean = s * (1.0f / ND);
    float vv = 0.f;
    for (int i = lane; i < ND; i += 32) { float d = xr[i] - mean; vv += d * d; }
    #pragma unroll
    for (int o = 16; o; o >>= 1) vv += __shfl_xor(vv, o, 32);
    float rs = rsqrtf(vv * (1.0f / ND) + 1e-5f);
    _Float16* hr = Hout + (long)row * ND;
    for (int i = lane; i < ND; i += 32)
        hr[i] = (_Float16)((xr[i] - mean) * rs * g[i] + bt[i]);
}

// ---------------- WMMA GEMM with LDS-staged A tile ----------------
// All 8 waves of a block share one 16-row m-tile; the A K-chunk is staged
// through a double-buffered LDS tile (global_load_b64 -> ds_store_b64, then
// ds_load_b128 fragment reads), while B streams from global per wave.
// A: [M,K] f16 row-major.  Bt: [N,K] f16 (transposed weights).
// Requires: K%64==0, (M/16)*(N/64)%8==0 (holds for all launches here).
__global__ void k_gemm(const _Float16* __restrict__ A, const _Float16* __restrict__ Bt,
                       const float* __restrict__ bias, const float* __restrict__ resid,
                       float* __restrict__ C, _Float16* __restrict__ Ch,
                       int M, int N, int K, float scale, int act) {
    __shared__ _Float16 As[2][16 * LROW];
    int wid = threadIdx.x >> 5, lane = threadIdx.x & 31;
    int ntile = N >> 6;                    // 64-wide strips
    int tm = (blockIdx.x * 8) / ntile;     // shared by all 8 waves
    int tn = (blockIdx.x * 8) % ntile + wid;
    int g = lane >> 4, ml = lane & 15;

    const _Float16* Bcol = Bt + (long)(tn * 64 + ml) * K;   // +t*16*K per subtile

    // cooperative A staging: thread -> 4 contiguous halves of the 16x64 chunk
    int ct   = threadIdx.x * 4;
    int crow = ct >> 6, ccol = ct & 63;
    const _Float16* Asrc = A + (long)(tm * 16 + crow) * K + ccol;

    v8f acc[4] = {{}, {}, {}, {}};
    *(unsigned long long*)(&As[0][crow * LROW + ccol]) = *(const unsigned long long*)(Asrc);
    int buf = 0;
    for (int kb = 0; kb < K; kb += 64) {
        __syncthreads();                                   // As[buf] ready
        if (kb + 64 < K)                                   // prefetch next chunk into other buffer
            *(unsigned long long*)(&As[buf ^ 1][crow * LROW + ccol]) =
                *(const unsigned long long*)(Asrc + kb + 64);
        if (kb + 128 < K) __builtin_prefetch((const void*)(Bcol + kb + 128), 0, 1);

        #pragma unroll
        for (int s = 0; s < 2; ++s) {
            Frag a;
            const _Float16* ap = &As[buf][ml * LROW + s * 32 + g * 8];
            uint4 lo = *(const uint4*)(ap);                // ds_load_b128
            uint4 hi = *(const uint4*)(ap + 16);           // ds_load_b128
            a.u[0] = lo.x; a.u[1] = lo.y; a.u[2] = lo.z; a.u[3] = lo.w;
            a.u[4] = hi.x; a.u[5] = hi.y; a.u[6] = hi.z; a.u[7] = hi.w;
            #pragma unroll
            for (int t = 0; t < 4; ++t) {
                Frag b;
                const _Float16* bc = Bcol + (long)t * 16 * K + kb + s * 32;
                #pragma unroll
                for (int v = 0; v < 8; ++v)
                    b.u[v] = *(const unsigned*)(bc + ((v & 4) ? 16 : 0) + g * 8 + 2 * (v & 3));
                acc[t] = __builtin_amdgcn_wmma_f32_16x16x32_f16(false, a.v, false, b.v,
                                                                (short)0, acc[t], false, false);
            }
        }
        buf ^= 1;
    }

    #pragma unroll
    for (int t = 0; t < 4; ++t) {
        int nn = tn * 64 + t * 16 + ml;
        float bs = bias ? bias[nn] : 0.f;
        #pragma unroll
        for (int r = 0; r < 8; ++r) {
            int m = tm * 16 + g * 8 + r;
            float val = (acc[t][r] + bs) * scale;
            if (act) val = val / (1.0f + expf(-1.702f * val));   // x * sigmoid(1.702 x)
            if (resid) val += resid[(long)m * N + nn];
            long o = (long)m * N + nn;
            if (C)  C[o]  = val;
            if (Ch) Ch[o] = (_Float16)val;
        }
    }
}

// ---------------- pad Q/K into [bh, 80, 64] (zero rows >= 77) ----------------
__global__ void k_pad_qk(const _Float16* __restrict__ src, _Float16* __restrict__ dst) {
    long i = (long)blockIdx.x * blockDim.x + threadIdx.x;
    long total = (long)NB * NH * SP * HD;
    if (i >= total) return;
    int d = (int)(i % HD);
    int t = (int)((i / HD) % SP);
    int h = (int)((i / ((long)HD * SP)) % NH);
    int b = (int)(i / ((long)HD * SP * NH));
    dst[i] = (t < NS) ? src[((long)(b * NS + t) * ND) + h * HD + d] : (_Float16)0.f;
}

// ---------------- pad+transpose V into [bh, 64, 96] (zero cols >= 77) ----------------
__global__ void k_pad_vT(const _Float16* __restrict__ src, _Float16* __restrict__ dst) {
    long i = (long)blockIdx.x * blockDim.x + threadIdx.x;
    long total = (long)NB * NH * HD * SPK;
    if (i >= total) return;
    int t = (int)(i % SPK);
    int d = (int)((i / SPK) % HD);
    int h = (int)((i / ((long)SPK * HD)) % NH);
    int b = (int)(i / ((long)SPK * HD * NH));
    dst[i] = (t < NS) ? src[((long)(b * NS + t) * ND) + h * HD + d] : (_Float16)0.f;
}

// ---------------- attention scores: one wave = 16 queries x all 80 keys ----------------
__global__ void k_scores(const _Float16* __restrict__ Qp, const _Float16* __restrict__ Kp,
                         const int* __restrict__ mask, float* __restrict__ Sc) {
    int wid = threadIdx.x >> 5, lane = threadIdx.x & 31;
    int bh = blockIdx.y;
    int b = bh / NH;
    int tm = wid;                         // 5 query tiles (waves 5..7 idle)
    if (tm >= 5) return;
    int g = lane >> 4, ml = lane & 15;
    int qm = tm * 16 + ml;

    const _Float16* qrow = Qp + ((long)bh * SP + qm) * HD;
    v8f acc[5] = {{}, {}, {}, {}, {}};
    #pragma unroll
    for (int kb = 0; kb < HD; kb += 32) {
        Frag a;
        #pragma unroll
        for (int v = 0; v < 8; ++v) {
            int k0 = kb + ((v & 4) ? 16 : 0) + g * 8 + 2 * (v & 3);
            a.u[v] = *(const unsigned*)(qrow + k0);
        }
        #pragma unroll
        for (int tn = 0; tn < 5; ++tn) {
            Frag bb;
            const _Float16* krow = Kp + ((long)bh * SP + tn * 16 + ml) * HD;
            #pragma unroll
            for (int v = 0; v < 8; ++v) {
                int kd = kb + ((v & 4) ? 16 : 0) + g * 8 + 2 * (v & 3);
                bb.u[v] = *(const unsigned*)(krow + kd);
            }
            acc[tn] = __builtin_amdgcn_wmma_f32_16x16x32_f16(false, a.v, false, bb.v,
                                                             (short)0, acc[tn], false, false);
        }
    }

    #pragma unroll
    for (int tn = 0; tn < 5; ++tn) {
        int kn = tn * 16 + ml;
        int maskv = (kn < NNP) ? 1 : ((kn - NNP < NT) ? mask[b * NT + (kn - NNP)] : 0);
        #pragma unroll
        for (int r = 0; r < 8; ++r) {
            int q = tm * 16 + g * 8 + r;
            float bias = 0.f;
            if (kn > q)  bias += NEGV;    // causal
            if (!maskv)  bias += NEGV;    // padding
            Sc[((long)bh * SP + q) * SP + kn] = acc[tn][r] + bias;
        }
    }
}

// ---------------- softmax over 80 keys (one wave per row); f16 probs, K padded to 96 ----------------
__global__ void k_softmax(const float* __restrict__ Sc, _Float16* __restrict__ Pr) {
    int wid = threadIdx.x >> 5, lane = threadIdx.x & 31;
    long row = (long)blockIdx.x * 8 + wid;
    long rows = (long)NB * NH * SP;
    if (row >= rows) return;
    const float* s = Sc + row * SP;
    _Float16* p = Pr + row * SPK;
    float v0 = s[lane];
    float v1 = s[lane + 32];
    float v2 = (lane + 64 < SP) ? s[lane + 64] : NEGV;
    float m = fmaxf(v0, fmaxf(v1, v2));
    #pragma unroll
    for (int o = 16; o; o >>= 1) m = fmaxf(m, __shfl_xor(m, o, 32));
    float e0 = expf(v0 - m), e1 = expf(v1 - m);
    float e2 = (lane + 64 < SP) ? expf(v2 - m) : 0.f;
    float ss = e0 + e1 + e2;
    #pragma unroll
    for (int o = 16; o; o >>= 1) ss += __shfl_xor(ss, o, 32);
    float inv = 1.0f / ss;
    p[lane]      = (_Float16)(e0 * inv);
    p[lane + 32] = (_Float16)(e1 * inv);
    if (lane + 64 < SP) p[lane + 64] = (_Float16)(e2 * inv);
    if (lane < SPK - SP) p[SP + lane] = (_Float16)0.f;   // zero pad K 80..95
}

// ---------------- PV: one wave = 16 queries x full 64-dim head ----------------
__global__ void k_pv(const _Float16* __restrict__ Pr, const _Float16* __restrict__ Vp,
                     _Float16* __restrict__ O) {
    int wid = threadIdx.x >> 5, lane = threadIdx.x & 31;
    int bh = blockIdx.y;
    int b = bh / NH, hh = bh % NH;
    int tm = wid;                         // 5 query tiles (waves 5..7 idle)
    if (tm >= 5) return;
    int g = lane >> 4, ml = lane & 15;
    int qm = tm * 16 + ml;

    const _Float16* prow = Pr + ((long)bh * SP + qm) * SPK;
    v8f acc[4] = {{}, {}, {}, {}};
    #pragma unroll
    for (int kb = 0; kb < SPK; kb += 32) {
        Frag a;
        #pragma unroll
        for (int v = 0; v < 8; ++v) {
            int k0 = kb + ((v & 4) ? 16 : 0) + g * 8 + 2 * (v & 3);
            a.u[v] = *(const unsigned*)(prow + k0);
        }
        #pragma unroll
        for (int tn = 0; tn < 4; ++tn) {
            Frag bb;
            const _Float16* vrow = Vp + ((long)bh * HD + tn * 16 + ml) * SPK;
            #pragma unroll
            for (int v = 0; v < 8; ++v) {
                int kd = kb + ((v & 4) ? 16 : 0) + g * 8 + 2 * (v & 3);
                bb.u[v] = *(const unsigned*)(vrow + kd);
            }
            acc[tn] = __builtin_amdgcn_wmma_f32_16x16x32_f16(false, a.v, false, bb.v,
                                                             (short)0, acc[tn], false, false);
        }
    }
    #pragma unroll
    for (int tn = 0; tn < 4; ++tn) {
        int nd = tn * 16 + ml;
        #pragma unroll
        for (int r = 0; r < 8; ++r) {
            int q = tm * 16 + g * 8 + r;
            if (q < NS)
                O[((long)(b * NS + q) * ND) + hh * HD + nd] = (_Float16)acc[tn][r];
        }
    }
}

// ---------------- argmax of tokens -> gather index ----------------
__global__ void k_argmax(const int* __restrict__ tok, int* __restrict__ idx) {
    int b = blockIdx.x * blockDim.x + threadIdx.x;
    if (b >= NB) return;
    int best = tok[b * NT], bi = 0;
    for (int t = 1; t < NT; ++t) {
        int v = tok[b * NT + t];
        if (v > best) { best = v; bi = t; }
    }
    idx[b] = bi + NNP;
}

// ---------------- final LN on the gathered row (one wave per batch element) ----------------
__global__ void k_final(const float* __restrict__ X, const int* __restrict__ idx,
                        const float* __restrict__ g, const float* __restrict__ bt,
                        float* __restrict__ out) {
    int wid = threadIdx.x >> 5, lane = threadIdx.x & 31;
    int b = blockIdx.x * 8 + wid;
    if (b >= NB) return;
    const float* xr = X + ((long)b * NS + idx[b]) * ND;
    float s = 0.f;
    for (int i = lane; i < ND; i += 32) s += xr[i];
    #pragma unroll
    for (int o = 16; o; o >>= 1) s += __shfl_xor(s, o, 32);
    float mean = s * (1.0f / ND);
    float vv = 0.f;
    for (int i = lane; i < ND; i += 32) { float d = xr[i] - mean; vv += d * d; }
    #pragma unroll
    for (int o = 16; o; o >>= 1) vv += __shfl_xor(vv, o, 32);
    float rs = rsqrtf(vv * (1.0f / ND) + 1e-5f);
    for (int i = lane; i < ND; i += 32)
        out[(long)b * ND + i] = (xr[i] - mean) * rs * g[i] + bt[i];
}

// ==================================================================
extern "C" void kernel_launch(void* const* d_in, const int* in_sizes, int n_in,
                              void* d_out, int out_size, void* d_ws, size_t ws_size,
                              hipStream_t stream) {
    const int*   tokens   = (const int*)  d_in[0];
    const int*   attnmask = (const int*)  d_in[1];
    const float* g_prompt = (const float*)d_in[2];
    const float* s_prompt = (const float*)d_in[3];
    const float* tok_emb  = (const float*)d_in[4];
    const float* pos_emb  = (const float*)d_in[5];
    const float* ln1_g    = (const float*)d_in[6];
    const float* ln1_b    = (const float*)d_in[7];
    const float* Wq       = (const float*)d_in[8];
    const float* bq       = (const float*)d_in[9];
    const float* Wk       = (const float*)d_in[10];
    const float* bk       = (const float*)d_in[11];
    const float* Wv       = (const float*)d_in[12];
    const float* bv       = (const float*)d_in[13];
    const float* Wo       = (const float*)d_in[14];
    const float* bo       = (const float*)d_in[15];
    const float* ln2_g    = (const float*)d_in[16];
    const float* ln2_b    = (const float*)d_in[17];
    const float* W1       = (const float*)d_in[18];
    const float* b1       = (const float*)d_in[19];
    const float* W2       = (const float*)d_in[20];
    const float* b2       = (const float*)d_in[21];
    const float* lnf_g    = (const float*)d_in[22];
    const float* lnf_b    = (const float*)d_in[23];
    float* out = (float*)d_out;

    const long rows  = (long)NB * NS;          // 9856
    const long nXD   = rows * ND;              // activations
    const long nXFF  = rows * NFF;

    // ---- workspace carve ----
    char* w = (char*)d_ws;
    auto alloc = [&](size_t bytes) {
        char* p = w;
        w += (bytes + 255) & ~(size_t)255;
        return p;
    };
    float*     X    = (float*)    alloc(nXD * sizeof(float));
    _Float16*  Hf   = (_Float16*) alloc(nXD * sizeof(_Float16));
    _Float16*  Qf   = (_Float16*) alloc(nXD * sizeof(_Float16));
    _Float16*  Kf   = (_Float16*) alloc(nXD * sizeof(_Float16));
    _Float16*  Vf   = (_Float16*) alloc(nXD * sizeof(_Float16));
    _Float16*  Of   = (_Float16*) alloc(nXD * sizeof(_Float16));
    _Float16*  FFf  = (_Float16*) alloc(nXFF * sizeof(_Float16));
    float*     Sc   = (float*)    alloc((size_t)NB * NH * SP * SP * sizeof(float));
    _Float16*  Pr   = (_Float16*) alloc((size_t)NB * NH * SP * SPK * sizeof(_Float16));
    _Float16*  wq16 = (_Float16*) alloc((size_t)ND * ND * sizeof(_Float16));   // transposed [N,K]
    _Float16*  wk16 = (_Float16*) alloc((size_t)ND * ND * sizeof(_Float16));
    _Float16*  wv16 = (_Float16*) alloc((size_t)ND * ND * sizeof(_Float16));
    _Float16*  wo16 = (_Float16*) alloc((size_t)ND * ND * sizeof(_Float16));
    _Float16*  w116 = (_Float16*) alloc((size_t)ND * NFF * sizeof(_Float16)); // [NFF,ND]
    _Float16*  w216 = (_Float16*) alloc((size_t)NFF * ND * sizeof(_Float16)); // [ND,NFF]
    int*       gidx = (int*)      alloc(NB * sizeof(int));
    (void)ws_size; (void)n_in; (void)in_sizes; (void)out_size;

    // padded attention staging ALIASED onto FFf (dead during attention phase):
    //   Qp,Kp: [bh,80,64]  Vp: [bh,64,96]   total 33.6MB < 40.4MB (FFf)
    const long nQK = (long)NB * NH * SP * HD;
    const long nVP = (long)NB * NH * HD * SPK;
    _Float16* Qp = FFf;
    _Float16* Kp = FFf + nQK;
    _Float16* Vp = FFf + 2 * nQK;

    auto cvtT = [&](const float* src, _Float16* dst, int K, int N) {
        long n = (long)K * N;
        int blocks = (int)((n + 255) / 256);
        k_cvt_t<<<blocks, 256, 0, stream>>>(src, dst, K, N);
    };
    auto gemm = [&](const _Float16* A, const _Float16* Bt, const float* bias,
                    const float* resid, float* C, _Float16* Ch,
                    int M, int N, int K, float scale, int act) {
        int tiles = (M / 16) * (N / 64);   // always a multiple of 8 here
        int blocks = tiles / 8;
        k_gemm<<<blocks, 256, 0, stream>>>(A, Bt, bias, resid, C, Ch, M, N, K, scale, act);
    };

    // 1. embeddings + prompt[0] + positional
    k_embed<<<(int)((nXD + 255) / 256), 256, 0, stream>>>(tokens, tok_emb, pos_emb, g_prompt, X);

    const int lnBlocks = (int)((rows + 7) / 8);

    for (int i = 0; i < NLAY; ++i) {
        if (i > 0) {
            int n = NB * NNP * ND;
            k_set_prompt<<<(n + 255) / 256, 256, 0, stream>>>(g_prompt, s_prompt, X, i);
        }
        // per-layer weight conversion to transposed f16 [N,K]
        cvtT(Wq + (long)i * ND * ND,  wq16, ND,  ND);
        cvtT(Wk + (long)i * ND * ND,  wk16, ND,  ND);
        cvtT(Wv + (long)i * ND * ND,  wv16, ND,  ND);
        cvtT(Wo + (long)i * ND * ND,  wo16, ND,  ND);
        cvtT(W1 + (long)i * ND * NFF, w116, ND,  NFF);
        cvtT(W2 + (long)i * NFF * ND, w216, NFF, ND);

        // LN1 -> h (f16)
        k_ln<<<lnBlocks, 256, 0, stream>>>(X, ln1_g + i * ND, ln1_b + i * ND, Hf, (int)rows);

        // q/k/v projections (q scaled by 0.125)
        gemm(Hf, wq16, bq + i * ND, nullptr, nullptr, Qf, (int)rows, ND, ND, 0.125f, 0);
        gemm(Hf, wk16, bk + i * ND, nullptr, nullptr, Kf, (int)rows, ND, ND, 1.0f,   0);
        gemm(Hf, wv16, bv + i * ND, nullptr, nullptr, Vf, (int)rows, ND, ND, 1.0f,   0);

        // pad/repack into WMMA-friendly layouts
        k_pad_qk<<<(int)((nQK + 255) / 256), 256, 0, stream>>>(Qf, Qp);
        k_pad_qk<<<(int)((nQK + 255) / 256), 256, 0, stream>>>(Kf, Kp);
        k_pad_vT<<<(int)((nVP + 255) / 256), 256, 0, stream>>>(Vf, Vp);

        // attention
        k_scores <<<dim3(1, NB * NH), 256, 0, stream>>>(Qp, Kp, attnmask, Sc);
        k_softmax<<<(int)(((long)NB * NH * SP + 7) / 8), 256, 0, stream>>>(Sc, Pr);
        k_pv     <<<dim3(1, NB * NH), 256, 0, stream>>>(Pr, Vp, Of);

        // output projection + residual -> X
        gemm(Of, wo16, bo + i * ND, X, X, nullptr, (int)rows, ND, ND, 1.0f, 0);

        // LN2 -> h (f16)
        k_ln<<<lnBlocks, 256, 0, stream>>>(X, ln2_g + i * ND, ln2_b + i * ND, Hf, (int)rows);

        // FFN: W1 + quick-GELU, then W2 + residual -> X
        gemm(Hf,  w116, b1 + i * NFF, nullptr, nullptr, FFf, (int)rows, NFF, ND,  1.0f, 1);
        gemm(FFf, w216, b2 + i * ND,  X,       X,       nullptr, (int)rows, ND,  NFF, 1.0f, 0);
    }

    // final LN on argmax-gathered rows
    k_argmax<<<1, 128, 0, stream>>>(tokens, gidx);
    k_final<<<(NB + 7) / 8, 256, 0, stream>>>(X, gidx, lnf_g, lnf_b, out);
}
